// AttentionLayer_62414464746063
// MI455X (gfx1250) — matrix operand order repaired and checked by reading, not compiled, not run on hardware
//
#include <hip/hip_runtime.h>

#define NH  12
#define SEQ 2048
#define DH  64
#define WID 768   // NH*DH

typedef __attribute__((ext_vector_type(16))) __bf16 v16bf;
typedef __attribute__((ext_vector_type(8)))  float  v8f;
typedef __attribute__((ext_vector_type(4)))  unsigned int v4u;
typedef __attribute__((ext_vector_type(8)))  int v8i;
typedef __attribute__((ext_vector_type(4)))  int v4i;

union Frag {
  unsigned int u[8];
  v16bf bf;
};

__device__ __forceinline__ unsigned short f2bf(float f) {
  unsigned int u = __float_as_uint(f);
  u += 0x7FFFu + ((u >> 16) & 1u);          // round-to-nearest-even
  return (unsigned short)(u >> 16);
}

__device__ __forceinline__ unsigned int pack2bf(float lo, float hi) {
  return (unsigned int)f2bf(lo) | ((unsigned int)f2bf(hi) << 16);
}

// ---- TDM: 2D tile load Global -> LDS (bf16 elements, data_size=2B)
// D# layout per CDNA5 ISA ch.8: group0 {count, lds_addr, global_addr, type=2},
// group1 {data_size, tensor dims/strides, tile dims}. Groups 2/3 zero (2D tile).
// This toolchain exposes the 6-arg builtin: (v4u, v8i, v4i, v4i, v8i, i32 cpol).
__device__ __forceinline__ void tdm_load_2d(unsigned lds_byte_addr,
                                            const void* gptr,
                                            unsigned tensor_d0, unsigned tensor_d1,
                                            unsigned stride0,
                                            unsigned tile_d0, unsigned tile_d1) {
  unsigned long long ga = (unsigned long long)(uintptr_t)gptr;
  v4u g0;
  g0[0] = 1u;                                             // count=1 (user descriptor)
  g0[1] = lds_byte_addr;                                  // lds_addr [63:32]
  g0[2] = (unsigned)(ga & 0xffffffffu);                   // global_addr lo
  g0[3] = (unsigned)((ga >> 32) & 0x1ffffffu) | (2u << 30); // global_addr hi | type=2
  v8i g1;
  g1[0] = (int)(1u << 16);                                // data_size=1 (2 bytes)
  g1[1] = (int)((tensor_d0 & 0xffffu) << 16);             // tensor_dim0[15:0] @ [63:48]
  g1[2] = (int)(((tensor_d0 >> 16) & 0xffffu) |           // tensor_dim0[31:16]
                ((tensor_d1 & 0xffffu) << 16));           // tensor_dim1[15:0]
  g1[3] = (int)(((tensor_d1 >> 16) & 0xffffu) |           // tensor_dim1[31:16]
                ((tile_d0 & 0xffffu) << 16));             // tile_dim0 @ [127:112]
  g1[4] = (int)(tile_d1 & 0xffffu);                       // tile_dim1; tile_dim2=0
  g1[5] = (int)stride0;                                   // tensor_dim0_stride lo32
  g1[6] = 0;                                              // stride0 hi16 | stride1 lo16
  g1[7] = 0;                                              // stride1 hi32
  v4i z4 = (v4i){0, 0, 0, 0};
  v8i z8 = (v8i){0, 0, 0, 0, 0, 0, 0, 0};
  __builtin_amdgcn_tensor_load_to_lds(g0, g1, z4, z4, z8, 0);
}

__device__ __forceinline__ unsigned lds_addr_of(const void* p) {
  // generic pointer to LDS: addr[31:0] is the LDS byte offset
  return (unsigned)(uintptr_t)p;
}

// ---------------------------------------------------------------- conversions

__global__ void xcvt_kernel(const float* __restrict__ x,
                            unsigned int* __restrict__ xb, int ndw) {
  int i = blockIdx.x * blockDim.x + threadIdx.x;
  if (i < ndw) xb[i] = pack2bf(x[2 * i], x[2 * i + 1]);
}

// W[k][n] (fp32) -> Wt[n][k] (bf16), tiled transpose through LDS
__global__ void __launch_bounds__(256) wcvt_kernel(const float* __restrict__ W,
                                                   unsigned short* __restrict__ Wt) {
  __shared__ float tile[32][33];
  const int tx = threadIdx.x & 31;
  const int ty = threadIdx.x >> 5;            // 0..7
  const int kb = blockIdx.x * 32;
  const int nb = blockIdx.y * 32;
  #pragma unroll
  for (int i = ty; i < 32; i += 8)
    tile[i][tx] = W[(size_t)(kb + i) * WID + nb + tx];
  __syncthreads();
  #pragma unroll
  for (int i = ty; i < 32; i += 8)
    Wt[(size_t)(nb + i) * WID + kb + tx] = f2bf(tile[tx][i]);
}

__global__ void cumsum_kernel(const int* __restrict__ seg, int* __restrict__ cs) {
  if (threadIdx.x == 0) {
    const int* s = seg + blockIdx.x * SEQ;
    int* o = cs + blockIdx.x * SEQ;
    int a = 0;
    for (int i = 0; i < SEQ; ++i) { a += s[i]; o[i] = a; }
  }
}

// ---------------------------------------------------------------- QKV GEMM

__device__ __forceinline__ void load_frag_a(Frag& f, const uint4* arow, int k8, int hf) {
  uint4 lo = arow[k8 + hf];          // K {0..7} / {8..15}
  uint4 hi = arow[k8 + hf + 2];      // K {16..23} / {24..31}
  *(uint4*)&f.u[0] = lo; *(uint4*)&f.u[4] = hi;
}
__device__ __forceinline__ void load_frag_b(Frag& f, const uint4* brow, int k8, int hf) {
  uint4 lo = brow[k8 + hf];          // K {0..7} / {8..15}
  uint4 hi = brow[k8 + hf + 2];      // K {16..23} / {24..31}
  *(uint4*)&f.u[0] = lo; *(uint4*)&f.u[4] = hi;
}

// One wave computes a 16x64 tile of out = x @ W + b; register double-buffered.
__global__ void __launch_bounds__(128) qkv_gemm(
    const unsigned int* __restrict__ xb,
    const unsigned int* __restrict__ wt,
    const float* __restrict__ bias,
    unsigned short* __restrict__ outp,
    const int vmode) {
  const int wave = threadIdx.x >> 5;
  const int lane = threadIdx.x & 31;
  const int hf = lane >> 4;
  const int nn = lane & 15;

  const int m0 = (blockIdx.x * 4 + wave) * 16;
  const int n0 = blockIdx.y * 64;

  v8f c[4];
  #pragma unroll
  for (int g = 0; g < 4; ++g) c[g] = (v8f){0.f,0.f,0.f,0.f,0.f,0.f,0.f,0.f};

  const uint4* arow = (const uint4*)(xb + (size_t)(m0 + nn) * 384);
  const uint4* brow[4];
  #pragma unroll
  for (int g = 0; g < 4; ++g)
    brow[g] = (const uint4*)(wt + (size_t)(n0 + 16 * g + nn) * 384);

  Frag a_c, b_c[4];
  load_frag_a(a_c, arow, 0, hf);
  #pragma unroll
  for (int g = 0; g < 4; ++g) load_frag_b(b_c[g], brow[g], 0, hf);

  for (int kk = 0; kk < WID; kk += 32) {
    const bool more = (kk + 32) < WID;
    Frag a_n, b_n[4];
    if (more) {                                   // prefetch next K-step
      const int k8 = (kk + 32) >> 3;
      load_frag_a(a_n, arow, k8, hf);
      #pragma unroll
      for (int g = 0; g < 4; ++g) load_frag_b(b_n[g], brow[g], k8, hf);
    }
    #pragma unroll
    for (int g = 0; g < 4; ++g)
      c[g] = __builtin_amdgcn_wmma_f32_16x16x32_bf16(
          false, a_c.bf, false, b_c[g].bf, (short)0, c[g], false, false);
    if (more) {
      a_c = a_n;
      #pragma unroll
      for (int g = 0; g < 4; ++g) b_c[g] = b_n[g];
    }
  }

  const int b = m0 / SEQ;
  #pragma unroll
  for (int g = 0; g < 4; ++g) {
    const int col = n0 + 16 * g + nn;
    const float bv = bias[col];
    const int h = col >> 6;
    const int d = col & 63;
    #pragma unroll
    for (int r = 0; r < 8; ++r) {
      const int m = m0 + r + 8 * hf;
      const int s = m & (SEQ - 1);
      const float v = c[g][r] + bv;
      if (vmode)
        outp[((size_t)((b * NH + h) * DH + d) << 11) + s] = f2bf(v);
      else
        outp[(((size_t)(b * NH + h) * SEQ + s) << 6) + d] = f2bf(v);
    }
  }
}

// ---------------------------------------------------------------- attention
// One wave per (b, h, 16-query tile). K/V chunks staged in LDS by the Tensor
// Data Mover, double-buffered per wave; waves are fully independent.
#define AW 2
__global__ void __launch_bounds__(64) attn_kernel(
    const unsigned int* __restrict__ Qb,      // bf16 [b][h][s][64] as dwords
    const unsigned int* __restrict__ Kb,      // bf16 [b][h][s][64] as dwords
    const unsigned int* __restrict__ Vt,      // bf16 [b][h][64][s] as dwords
    const int* __restrict__ cs,
    float* __restrict__ out) {
  __shared__ __align__(16) unsigned int kbuf[AW][2][1024]; // 32 keys x 64 halfs
  __shared__ __align__(16) unsigned int vbuf[AW][2][1024]; // 64 d-rows x 32 keys
  __shared__ __align__(16) unsigned int plds[AW][256];     // P tile 16x32 bf16

  const int wave = threadIdx.x >> 5;
  const int lane = threadIdx.x & 31;
  const int hf = lane >> 4;
  const int nn = lane & 15;

  const int flat = blockIdx.x * AW + wave;    // (b*NH + h)*128 + qtile
  const int qt = flat & 127;
  const int bh = flat >> 7;
  const int h = bh % NH;
  const int b = bh / NH;
  const int q0 = qt << 4;

  const unsigned int* qhead = Qb + (size_t)(b * NH + h) * SEQ * 32;
  const char* kbytes = (const char*)(Kb + (size_t)(b * NH + h) * SEQ * 32);
  const char* vbytes = (const char*)(Vt + (size_t)(b * NH + h) * DH * 1024);
  const int* csb = cs + b * SEQ;

  // Q fragments (two K-halves of d=64), loaded once from global
  Frag qf[2];
  {
    const uint4* qrow = (const uint4*)(qhead + (size_t)(q0 + nn) * 32);
    #pragma unroll
    for (int hh = 0; hh < 2; ++hh) {
      uint4 lo = qrow[4 * hh + hf];
      uint4 hi = qrow[4 * hh + hf + 2];
      *(uint4*)&qf[hh].u[0] = lo; *(uint4*)&qf[hh].u[4] = hi;
    }
  }

  int csq[8];
  #pragma unroll
  for (int r = 0; r < 8; ++r) csq[r] = csb[q0 + r + 8 * hf];

  float mrun[8], lrun[8];
  v8f acc[4];
  #pragma unroll
  for (int r = 0; r < 8; ++r) { mrun[r] = -1e30f; lrun[r] = 0.f; }
  #pragma unroll
  for (int g = 0; g < 4; ++g) acc[g] = (v8f){0.f,0.f,0.f,0.f,0.f,0.f,0.f,0.f};

  // TDM prefetch of a 32-key chunk into LDS buffer `bb`
  auto issueKV = [&](int key0, int bb) {
    // K tile: rows = 32 keys, row = 64 halfs contiguous
    tdm_load_2d(lds_addr_of(&kbuf[wave][bb][0]), kbytes + (size_t)key0 * DH * 2,
                /*tensor_d0=*/DH, /*tensor_d1=*/SEQ, /*stride0=*/DH,
                /*tile_d0=*/DH, /*tile_d1=*/32);
    // V tile: rows = 64 d-rows, 32 contiguous keys each, row stride = SEQ halfs
    tdm_load_2d(lds_addr_of(&vbuf[wave][bb][0]), vbytes + (size_t)key0 * 2,
                /*tensor_d0=*/SEQ, /*tensor_d1=*/DH, /*stride0=*/SEQ,
                /*tile_d0=*/32, /*tile_d1=*/DH);
  };
  issueKV(0, 0);
  issueKV(32, 1);

  for (int key0 = 0; key0 < SEQ; key0 += 32) {
    const int bb = (key0 >> 5) & 1;
    // wait for this buffer's pair; next chunk's pair (2 ops) may stay in flight
    if (key0 + 32 < SEQ) __builtin_amdgcn_s_wait_tensorcnt(2);
    else                 __builtin_amdgcn_s_wait_tensorcnt(0);

    // ---- scores: two 16-key subtiles, K=64 split into two WMMAs (B from LDS)
    v8f sc[2];
    #pragma unroll
    for (int t = 0; t < 2; ++t) {
      v8f cc = (v8f){0.f,0.f,0.f,0.f,0.f,0.f,0.f,0.f};
      const uint4* krow = (const uint4*)&kbuf[wave][bb][(16 * t + nn) * 32];
      #pragma unroll
      for (int hh = 0; hh < 2; ++hh) {
        Frag kf;
        uint4 lo = krow[4 * hh + hf];
        uint4 hi = krow[4 * hh + hf + 2];
        *(uint4*)&kf.u[0] = lo; *(uint4*)&kf.u[4] = hi;
        cc = __builtin_amdgcn_wmma_f32_16x16x32_bf16(
            false, qf[hh].bf, false, kf.bf, (short)0, cc, false, false);
      }
      sc[t] = cc;
    }

    const int ck0 = csb[key0 + nn];
    const int ck1 = csb[key0 + 16 + nn];

    // ---- online softmax (row spans one 16-lane group in the C layout)
    float p0[8], p1[8];
    #pragma unroll
    for (int r = 0; r < 8; ++r) {
      const int ci = csq[r];
      const float x0 = sc[0][r] * 0.125f;
      const float x1 = sc[1][r] * 0.125f;
      const bool k0m = (ck0 > ci);
      const bool k1m = (ck1 > ci);
      float rm = fmaxf(k0m ? -1e30f : x0, k1m ? -1e30f : x1);
      #pragma unroll
      for (int mm = 1; mm <= 8; mm <<= 1)
        rm = fmaxf(rm, __shfl_xor(rm, mm, 32));
      const float mn = fmaxf(mrun[r], rm);
      p0[r] = k0m ? 0.f : __expf(x0 - mn);
      p1[r] = k1m ? 0.f : __expf(x1 - mn);
      float rs = p0[r] + p1[r];
      #pragma unroll
      for (int mm = 1; mm <= 8; mm <<= 1)
        rs += __shfl_xor(rs, mm, 32);
      const float fr = __expf(mrun[r] - mn);
      lrun[r] = lrun[r] * fr + rs;
      mrun[r] = mn;
      #pragma unroll
      for (int g = 0; g < 4; ++g) acc[g][r] *= fr;
    }

    // ---- C-layout -> A-layout transpose of P through per-wave LDS slab
    unsigned short* pl = (unsigned short*)plds[wave];
    #pragma unroll
    for (int r = 0; r < 8; ++r) {
      const int M = r + 8 * hf;
      pl[M * 32 + nn]      = f2bf(p0[r]);
      pl[M * 32 + 16 + nn] = f2bf(p1[r]);
    }
    // intra-wave DS ordering: stores complete before cross-lane reads
    asm volatile("s_wait_dscnt 0x0" ::: "memory");

    Frag pf;
    {
      const uint4* prow = (const uint4*)&plds[wave][nn * 16];
      *(uint4*)&pf.u[0] = prow[hf];
      *(uint4*)&pf.u[4] = prow[hf + 2];
    }

    // ---- ctx += P @ V (B fragments from LDS-staged transposed V)
    #pragma unroll
    for (int g = 0; g < 4; ++g) {
      const uint4* vrow = (const uint4*)&vbuf[wave][bb][(16 * g + nn) * 16];
      Frag vf;
      uint4 lo = vrow[hf];
      uint4 hi = vrow[hf + 2];
      *(uint4*)&vf.u[0] = lo; *(uint4*)&vf.u[4] = hi;
      acc[g] = __builtin_amdgcn_wmma_f32_16x16x32_bf16(
          false, pf.bf, false, vf.bf, (short)0, acc[g], false, false);
    }

    // ---- prefetch chunk key0+64 into this buffer once its DS reads are done
    if (key0 + 64 < SEQ) {
      asm volatile("s_wait_dscnt 0x0" ::: "memory");
      issueKV(key0 + 64, bb);
    }
  }

  // ---- epilogue: normalize and write fp32 [b][s][768]
  #pragma unroll
  for (int g = 0; g < 4; ++g) {
    #pragma unroll
    for (int r = 0; r < 8; ++r) {
      const int s = q0 + r + 8 * hf;
      const int col = h * DH + 16 * g + nn;
      out[(size_t)(b * SEQ + s) * WID + col] = acc[g][r] / lrun[r];
    }
  }
}

// ---------------------------------------------------------------- launch

extern "C" void kernel_launch(void* const* d_in, const int* in_sizes, int n_in,
                              void* d_out, int out_size, void* d_ws, size_t ws_size,
                              hipStream_t stream) {
  (void)in_sizes; (void)n_in; (void)out_size; (void)ws_size;
  const float* x  = (const float*)d_in[0];
  const int*  seg = (const int*)d_in[1];
  const float* Wq = (const float*)d_in[2];
  const float* bq = (const float*)d_in[3];
  const float* Wk = (const float*)d_in[4];
  const float* bk = (const float*)d_in[5];
  const float* Wv = (const float*)d_in[6];
  const float* bv = (const float*)d_in[7];
  float* out = (float*)d_out;

  char* ws = (char*)d_ws;
  size_t off = 0;
  auto take = [&](size_t bytes) -> char* {
    char* p = ws + off;
    off += (bytes + 255) & ~(size_t)255;
    return p;
  };
  unsigned int*   xb  = (unsigned int*)  take((size_t)4 * SEQ * WID * 2);
  unsigned short* wtq = (unsigned short*)take((size_t)WID * WID * 2);
  unsigned short* wtk = (unsigned short*)take((size_t)WID * WID * 2);
  unsigned short* wtv = (unsigned short*)take((size_t)WID * WID * 2);
  unsigned short* qb  = (unsigned short*)take((size_t)4 * NH * SEQ * DH * 2);
  unsigned short* kb  = (unsigned short*)take((size_t)4 * NH * SEQ * DH * 2);
  unsigned short* vt  = (unsigned short*)take((size_t)4 * NH * SEQ * DH * 2);
  int*            cs  = (int*)           take((size_t)4 * SEQ * 4);

  const int ndw = 4 * SEQ * WID / 2;
  xcvt_kernel<<<(ndw + 255) / 256, 256, 0, stream>>>(x, xb, ndw);
  dim3 wg(WID / 32, WID / 32);
  wcvt_kernel<<<wg, 256, 0, stream>>>(Wq, wtq);
  wcvt_kernel<<<wg, 256, 0, stream>>>(Wk, wtk);
  wcvt_kernel<<<wg, 256, 0, stream>>>(Wv, wtv);
  cumsum_kernel<<<4, 32, 0, stream>>>(seg, cs);

  dim3 gg(4 * SEQ / 64, WID / 64);   // (128, 12)
  qkv_gemm<<<gg, 128, 0, stream>>>(xb, (const unsigned int*)wtq, bq, qb, 0);
  qkv_gemm<<<gg, 128, 0, stream>>>(xb, (const unsigned int*)wtk, bk, kb, 0);
  qkv_gemm<<<gg, 128, 0, stream>>>(xb, (const unsigned int*)wtv, bv, vt, 1);

  const int nflat = 4 * NH * (SEQ / 16);   // 6144 wave-tiles
  attn_kernel<<<nflat / AW, 64, 0, stream>>>(
      (const unsigned int*)qb, (const unsigned int*)kb,
      (const unsigned int*)vt, cs, out);
}
